// Block_73495480369596
// MI455X (gfx1250) — matrix operand order repaired and checked by reading, not compiled, not run on hardware
//
#include <hip/hip_runtime.h>
#include <hip/hip_bf16.h>

// ---------------------------------------------------------------------------
// Types for CDNA5 WMMA
// ---------------------------------------------------------------------------
typedef __attribute__((ext_vector_type(16))) __bf16 v16bf;
typedef __attribute__((ext_vector_type(8)))  float  v8f;

union FragBF {
    v16bf        v;
    unsigned int u[8];
};

// f32 -> bf16 (round-to-nearest, ties away: one VALU op)
__device__ __forceinline__ unsigned short f2bf_raw(float f) {
    unsigned u = __builtin_bit_cast(unsigned, f);
    return (unsigned short)((u + 0x8000u) >> 16);
}

// pack two floats into a bf16 pair [b|a] via v_perm_b32: 3 VALU total
__device__ __forceinline__ unsigned pack_bf16(float a, float b) {
    unsigned ua = __builtin_bit_cast(unsigned, a) + 0x8000u;
    unsigned ub = __builtin_bit_cast(unsigned, b) + 0x8000u;
    // D bytes: [3]=ub.b3 [2]=ub.b2 [1]=ua.b3 [0]=ua.b2  (sel 4..7 -> S0, 0..3 -> S1)
    return __builtin_amdgcn_perm(ub, ua, 0x07060302u);
}

__device__ __forceinline__ float gelu_f(float x) {
    return 0.5f * x * (1.0f + erff(x * 0.7071067811865475f));
}

#define BK 32

// ---------------------------------------------------------------------------
// Templated GEMM:  C[m][n] = sum_k A[m, acol0+k] * W[n, k]  (+bias, +residual)
// 256 thr = 8 waves. WVM x WVN wave grid, each wave WM x WN 16x16 WMMA tiles.
// Block tile (WVM*WM*16) x (WVN*WN*16). Double-buffered bf16 LDS staging,
// float4 global loads, v_perm packed bf16 stores.
// ---------------------------------------------------------------------------
template <int WM, int WN, int WVM, int WVN>
__global__ __launch_bounds__(256)
void gemm_wmma(const float* __restrict__ A, int lda, int acol0,
               const float* __restrict__ W,
               const float* __restrict__ bias,            // nullable
               const float* __restrict__ res, int ldres,  // nullable
               float* __restrict__ Cout, int ldc, int ccol0,
               int M, int Nc, int K)
{
    constexpr int BMt = WVM * WM * 16;
    constexpr int BNt = WVN * WN * 16;

    __shared__ __align__(16) unsigned short As[2][BMt][BK + 2];
    __shared__ __align__(16) unsigned short Bs[2][BNt][BK + 2];

    const int t     = threadIdx.x;
    const int lane  = t & 31;
    const int wv    = t >> 5;
    const int lo    = lane & 15;
    const int hi    = lane >> 4;
    const int waveM = wv / WVN;
    const int waveN = wv % WVN;
    const int bm0   = blockIdx.y * BMt;
    const int bn0   = blockIdx.x * BNt;

    v8f acc[WM][WN];
#pragma unroll
    for (int i = 0; i < WM; ++i)
#pragma unroll
        for (int j = 0; j < WN; ++j)
            acc[i][j] = (v8f){0.f,0.f,0.f,0.f,0.f,0.f,0.f,0.f};

    int kA[8], kB[8];
#pragma unroll
    for (int p = 0; p < 8; ++p) {
        kA[p] = hi * 8 + ((p < 4) ? 2 * p : 16 + 2 * (p - 4));
        kB[p] = hi * 8 + ((p < 4) ? 2 * p : 16 + 2 * (p - 4));
    }

    auto loadTiles = [&](int k0, int sel) {
        // A tile: BMt x 32 f32 -> BMt*8 float4, 256 thr
#pragma unroll
        for (int i = 0; i < BMt / 32; ++i) {
            int idx4 = t + i * 256;
            int r    = idx4 >> 3;            // 8 float4 per row
            int c4   = (idx4 & 7) * 4;
            int gm   = bm0 + r;
            if (gm >= M) gm = M - 1;
            float4 v = *(const float4*)&A[(long)gm * lda + acol0 + k0 + c4];
            *(unsigned*)&As[sel][r][c4]     = pack_bf16(v.x, v.y);
            *(unsigned*)&As[sel][r][c4 + 2] = pack_bf16(v.z, v.w);
        }
        // W tile: BNt x 32 f32
#pragma unroll
        for (int i = 0; i < BNt / 32; ++i) {
            int idx4 = t + i * 256;
            int r    = idx4 >> 3;
            int c4   = (idx4 & 7) * 4;
            float4 v = *(const float4*)&W[(long)(bn0 + r) * K + k0 + c4];
            *(unsigned*)&Bs[sel][r][c4]     = pack_bf16(v.x, v.y);
            *(unsigned*)&Bs[sel][r][c4 + 2] = pack_bf16(v.z, v.w);
        }
    };

    const int nk = K / BK;
    loadTiles(0, 0);
    __syncthreads();

    int sel = 0;
    for (int kt = 0; kt < nk; ++kt) {
        if (kt + 1 < nk) loadTiles((kt + 1) * BK, sel ^ 1);
        if (kt + 2 < nk) {     // L2 prefetch of the tile after next
            int gm = bm0 + (t >> 5);
            if (gm >= M) gm = M - 1;
            __builtin_prefetch(&A[(long)gm * lda + acol0 + (kt + 2) * BK + (t & 31)], 0, 1);
        }

        FragBF a[WM], b[WN];
#pragma unroll
        for (int i = 0; i < WM; ++i) {
            int row = waveM * WM * 16 + i * 16 + lo;
#pragma unroll
            for (int p = 0; p < 8; ++p)
                a[i].u[p] = *(const unsigned int*)&As[sel][row][kA[p]];
        }
#pragma unroll
        for (int j = 0; j < WN; ++j) {
            int col = waveN * WN * 16 + j * 16 + lo;
#pragma unroll
            for (int p = 0; p < 8; ++p)
                b[j].u[p] = *(const unsigned int*)&Bs[sel][col][kB[p]];
        }
#pragma unroll
        for (int i = 0; i < WM; ++i)
#pragma unroll
            for (int j = 0; j < WN; ++j)
                acc[i][j] = __builtin_amdgcn_wmma_f32_16x16x32_bf16(
                    false, a[i].v, false, b[j].v, (short)0, acc[i][j], false, false);
        __syncthreads();
        sel ^= 1;
    }

#pragma unroll
    for (int i = 0; i < WM; ++i) {
#pragma unroll
        for (int j = 0; j < WN; ++j) {
            int gn = bn0 + waveN * WN * 16 + j * 16 + lo;
#pragma unroll
            for (int r = 0; r < 8; ++r) {
                int gm = bm0 + waveM * WM * 16 + i * 16 + hi * 8 + r;
                if (gm < M) {
                    float val = acc[i][j][r];
                    if (bias) val += bias[gn];
                    if (res)  val += res[(long)gm * ldres + gn];
                    Cout[(long)gm * ldc + ccol0 + gn] = val;
                }
            }
        }
    }
}

// ---------------------------------------------------------------------------
// Dense 3x3 conv 64->64 as implicit GEMM on WMMA.
// K = 9 segments x 64 channels; segment applies shift (dy,dx) w/ zero pad.
// ---------------------------------------------------------------------------
#define CBM 128
#define CBN 64

__global__ __launch_bounds__(256)
void conv3x3_gemm_wmma(const float* __restrict__ Y,
                       const float* __restrict__ Wc,    // (64,64,3,3)
                       const float* __restrict__ bias,
                       float* __restrict__ CP,
                       int Bc, int Hh, int Ww)
{
    const int Npix = Hh * Ww;
    const int M    = Bc * Npix;

    __shared__ __align__(16) unsigned short As[2][CBM][BK + 2];
    __shared__ __align__(16) unsigned short Bs[2][CBN][BK + 2];

    const int t     = threadIdx.x;
    const int lane  = t & 31;
    const int wv    = t >> 5;
    const int lo    = lane & 15;
    const int hi    = lane >> 4;
    const int waveM = wv >> 1;
    const int waveN = wv & 1;
    const int bm0   = blockIdx.y * CBM;   // bn0 == 0 (Nc = 64)

    v8f c00 = {0.f,0.f,0.f,0.f,0.f,0.f,0.f,0.f};
    v8f c01 = c00, c10 = c00, c11 = c00;

    int kA[8], kB[8];
#pragma unroll
    for (int p = 0; p < 8; ++p) {
        kA[p] = hi * 8 + ((p < 4) ? 2 * p : 16 + 2 * (p - 4));
        kB[p] = hi * 8 + ((p < 4) ? 2 * p : 16 + 2 * (p - 4));
    }

    auto loadTiles = [&](int kt, int sel) {
        int k0  = kt * BK;
        int seg = k0 >> 6;          // 0..8 -> (ky,kx)
        int kk0 = k0 & 63;          // 0 or 32
        int dy  = seg / 3 - 1;
        int dx  = seg % 3 - 1;
#pragma unroll
        for (int i = 0; i < 4; ++i) {
            int idx4 = t + i * 256;
            int r    = idx4 >> 3;
            int c4   = (idx4 & 7) * 4;
            int gm   = bm0 + r;
            if (gm >= M) gm = M - 1;
            int b   = gm / Npix;
            int rem = gm - b * Npix;
            int oy  = rem / Ww;
            int ox  = rem - oy * Ww;
            int iy  = oy + dy, ix = ox + dx;
            unsigned u01 = 0u, u23 = 0u;
            if (iy >= 0 && iy < Hh && ix >= 0 && ix < Ww) {
                float4 v = *(const float4*)&Y[((long)b * Npix + iy * Ww + ix) * 256 + kk0 + c4];
                u01 = pack_bf16(v.x, v.y);
                u23 = pack_bf16(v.z, v.w);
            }
            *(unsigned*)&As[sel][r][c4]     = u01;
            *(unsigned*)&As[sel][r][c4 + 2] = u23;
        }
        // W tile: 64x32, w[n][k] = Wc[(n*64 + kk0+col)*9 + seg]
#pragma unroll
        for (int i = 0; i < 8; ++i) {
            int idx = t + i * 256;
            int r   = idx >> 5;
            int col = idx & 31;
            Bs[sel][r][col] = f2bf_raw(Wc[((long)r * 64 + kk0 + col) * 9 + seg]);
        }
    };

    const int nk = (9 * 64) / BK;   // 18
    loadTiles(0, 0);
    __syncthreads();

    int sel = 0;
    for (int kt = 0; kt < nk; ++kt) {
        if (kt + 1 < nk) loadTiles(kt + 1, sel ^ 1);

        FragBF a0, a1, b0, b1;
#pragma unroll
        for (int p = 0; p < 8; ++p) {
            a0.u[p] = *(const unsigned int*)&As[sel][waveM * 32 + lo][kA[p]];
            a1.u[p] = *(const unsigned int*)&As[sel][waveM * 32 + 16 + lo][kA[p]];
            b0.u[p] = *(const unsigned int*)&Bs[sel][waveN * 32 + lo][kB[p]];
            b1.u[p] = *(const unsigned int*)&Bs[sel][waveN * 32 + 16 + lo][kB[p]];
        }
        c00 = __builtin_amdgcn_wmma_f32_16x16x32_bf16(false, a0.v, false, b0.v, (short)0, c00, false, false);
        c01 = __builtin_amdgcn_wmma_f32_16x16x32_bf16(false, a0.v, false, b1.v, (short)0, c01, false, false);
        c10 = __builtin_amdgcn_wmma_f32_16x16x32_bf16(false, a1.v, false, b0.v, (short)0, c10, false, false);
        c11 = __builtin_amdgcn_wmma_f32_16x16x32_bf16(false, a1.v, false, b1.v, (short)0, c11, false, false);
        __syncthreads();
        sel ^= 1;
    }

    auto storeTile = [&](const v8f& c, int mt, int nt) {
        int gn = nt + lo;
#pragma unroll
        for (int r = 0; r < 8; ++r) {
            int gm = bm0 + mt + hi * 8 + r;
            if (gm < M) CP[(long)gm * 64 + gn] = c[r] + bias[gn];
        }
    };
    storeTile(c00, waveM * 32,      waveN * 32);
    storeTile(c01, waveM * 32,      waveN * 32 + 16);
    storeTile(c10, waveM * 32 + 16, waveN * 32);
    storeTile(c11, waveM * 32 + 16, waveN * 32 + 16);
}

// ---------------------------------------------------------------------------
// LayerNorm over a row of width Cw (32 | Cw), one wave per row. Optional GELU.
// ---------------------------------------------------------------------------
__global__ void ln_kernel(const float* __restrict__ in, int ldin, int incol,
                          const float* __restrict__ g, const float* __restrict__ bta,
                          float* __restrict__ out, int ldout, int outcol,
                          int Cw, int rows, int do_gelu)
{
    int lane = threadIdx.x & 31;
    int row  = blockIdx.x * (blockDim.x >> 5) + (threadIdx.x >> 5);
    if (row >= rows) return;
    int per = Cw >> 5;

    const float* ip = in + (long)row * ldin + incol;
    float vals[8];
    float s = 0.f;
#pragma unroll
    for (int j = 0; j < 8; ++j) {
        vals[j] = 0.f;
        if (j < per) { vals[j] = ip[lane + j * 32]; s += vals[j]; }
    }
    for (int o = 16; o; o >>= 1) s += __shfl_xor(s, o, 32);
    float mean = s / (float)Cw;

    float sq = 0.f;
#pragma unroll
    for (int j = 0; j < 8; ++j)
        if (j < per) { float d = vals[j] - mean; sq += d * d; }
    for (int o = 16; o; o >>= 1) sq += __shfl_xor(sq, o, 32);
    float inv = rsqrtf(sq / (float)Cw + 1e-5f);

    float* op = out + (long)row * ldout + outcol;
#pragma unroll
    for (int j = 0; j < 8; ++j) {
        if (j < per) {
            int c = lane + j * 32;
            float y = (vals[j] - mean) * inv * g[c] + bta[c];
            if (do_gelu) y = gelu_f(y);
            op[c] = y;
        }
    }
}

// ---------------------------------------------------------------------------
// Depthwise conv (data stored as (b, h*W+w, c) rows). Optional residual/GELU.
// ---------------------------------------------------------------------------
__global__ void dwconv_kernel(const float* __restrict__ in, int ldin, int incol,
                              const float* __restrict__ wgt, const float* __restrict__ bias,
                              float* __restrict__ out, int ldout, int outcol,
                              int Bc, int Cn, int Hh, int Ww, int ks, int stride, int pad,
                              int OH, int OW, int add_residual, int do_gelu)
{
    long idx   = (long)blockIdx.x * blockDim.x + threadIdx.x;
    long total = (long)Bc * OH * OW * Cn;
    if (idx >= total) return;
    int c  = (int)(idx % Cn); long t = idx / Cn;
    int ox = (int)(t % OW);   t /= OW;
    int oy = (int)(t % OH);
    int b  = (int)(t / OH);

    const float* ib = in + ((long)b * Hh * Ww) * ldin + incol;
    const float* wc = wgt + (long)c * ks * ks;
    float acc = bias[c];
    for (int ky = 0; ky < ks; ++ky) {
        int iy = oy * stride - pad + ky;
        if (iy < 0 || iy >= Hh) continue;
        for (int kx = 0; kx < ks; ++kx) {
            int ix = ox * stride - pad + kx;
            if (ix < 0 || ix >= Ww) continue;
            acc += wc[ky * ks + kx] * ib[(long)(iy * Ww + ix) * ldin + c];
        }
    }
    if (add_residual) acc += ib[(long)(oy * Ww + ox) * ldin + c];
    if (do_gelu)      acc = gelu_f(acc);
    out[((long)b * OH * OW + (long)oy * OW + ox) * ldout + outcol + c] = acc;
}

// ---------------------------------------------------------------------------
// Attention: per-thread online-softmax over Mkv keys, hd=24, 4 heads (grid.y).
// ---------------------------------------------------------------------------
__global__ void attn_kernel(const float* __restrict__ Q, const float* __restrict__ KV,
                            float* __restrict__ yhat,
                            int Bc, int Nn, int Mkv, int qcol_base, int outcol_base)
{
    long row = (long)blockIdx.x * blockDim.x + threadIdx.x;
    if (row >= (long)Bc * Nn) return;
    int h = blockIdx.y;
    int b = (int)(row / Nn);

    const float* qp = Q + row * 192 + qcol_base + h * 24;
    float q[24];
#pragma unroll
    for (int d = 0; d < 24; ++d) q[d] = qp[d];

    const float scale = 0.2041241452319315f;  // 24^-0.5
    float acc[24];
#pragma unroll
    for (int d = 0; d < 24; ++d) acc[d] = 0.f;
    float mrun = -1e30f, lrun = 0.f;

    const float* kvb = KV + (long)b * Mkv * 192;
    for (int m = 0; m < Mkv; ++m) {
        const float* kp = kvb + (long)m * 192 + h * 24;
        float s = 0.f;
#pragma unroll
        for (int d = 0; d < 24; ++d) s += q[d] * kp[d];
        s *= scale;
        float mn   = fmaxf(mrun, s);
        float corr = __expf(mrun - mn);
        float p    = __expf(s - mn);
        lrun = lrun * corr + p;
        const float* vp = kp + 96;
#pragma unroll
        for (int d = 0; d < 24; ++d) acc[d] = acc[d] * corr + p * vp[d];
        mrun = mn;
    }
    float inv = 1.f / lrun;
    float* op = yhat + row * 256 + outcol_base + h * 24;
#pragma unroll
    for (int d = 0; d < 24; ++d) op[d] = acc[d] * inv;
}

// ---------------------------------------------------------------------------
// Host orchestration
// ---------------------------------------------------------------------------
extern "C" void kernel_launch(void* const* d_in, const int* in_sizes, int n_in,
                              void* d_out, int out_size, void* d_ws, size_t ws_size,
                              hipStream_t stream)
{
    const int Hh = 56, Ww = 56, C = 256, N = Hh * Ww;
    const int B  = in_sizes[0] / (N * C);
    const long M = (long)B * N;

    const float* x        = (const float*)d_in[0];
    const float* lpu_w    = (const float*)d_in[3];
    const float* lpu_b    = (const float*)d_in[4];
    const float* n1_g     = (const float*)d_in[5];
    const float* n1_b     = (const float*)d_in[6];
    const float* conv2d_w = (const float*)d_in[7];
    const float* conv2d_b = (const float*)d_in[8];
    const float* cn_g     = (const float*)d_in[9];
    const float* cn_b     = (const float*)d_in[10];
    const float* q_w      = (const float*)d_in[11];
    const float* sr1_w    = (const float*)d_in[12];
    const float* sr1_b    = (const float*)d_in[13];
    const float* an1_g    = (const float*)d_in[14];
    const float* an1_b    = (const float*)d_in[15];
    const float* sr2_w    = (const float*)d_in[16];
    const float* sr2_b    = (const float*)d_in[17];
    const float* an2_g    = (const float*)d_in[18];
    const float* an2_b    = (const float*)d_in[19];
    const float* kv1_w    = (const float*)d_in[20];
    const float* kv2_w    = (const float*)d_in[21];
    const float* proj_w   = (const float*)d_in[22];
    const float* proj_b   = (const float*)d_in[23];
    const float* n2_g     = (const float*)d_in[24];
    const float* n2_b     = (const float*)d_in[25];
    const float* fc1_w    = (const float*)d_in[26];
    const float* fc1_b    = (const float*)d_in[27];
    const float* dw_w     = (const float*)d_in[28];
    const float* dw_b     = (const float*)d_in[29];
    const float* fc2_w    = (const float*)d_in[30];
    const float* fc2_b    = (const float*)d_in[31];
    float* out = (float*)d_out;

    // Workspace layout (floats)
    float* ws = (float*)d_ws;
    size_t off = 0;
    float* X    = ws + off; off += (size_t)M * 256;
    float* Y    = ws + off; off += (size_t)M * 256;
    float* YHAT = ws + off; off += (size_t)M * 256;
    float* XR   = ws + off; off += (size_t)M * 256;
    float* Qb   = ws + off; off += (size_t)M * 192;
    float* CP   = ws + off; off += (size_t)M * 64;
    float* X1   = ws + off; off += (size_t)B * 49  * 192;
    float* X2   = ws + off; off += (size_t)B * 196 * 192;
    float* KV1  = ws + off; off += (size_t)B * 49  * 192;
    float* KV2  = ws + off; off += (size_t)B * 196 * 192;
    float* Hb   = ws + off; off += (size_t)M * 1024;
    float* Gb   = ws + off; off += (size_t)M * 1024;
    (void)ws_size; (void)n_in; (void)out_size;

    // 1) LPU: depthwise 3x3 + residual -> X
    {
        long tot = M * 256;
        dwconv_kernel<<<(int)((tot + 255) / 256), 256, 0, stream>>>(
            x, 256, 0, lpu_w, lpu_b, X, 256, 0,
            B, 256, Hh, Ww, 3, 1, 1, Hh, Ww, 1, 0);
    }
    // 2) LayerNorm n1 -> Y
    ln_kernel<<<(int)((M + 3) / 4), 128, 0, stream>>>(X, 256, 0, n1_g, n1_b,
                                                      Y, 256, 0, 256, (int)M, 0);
    // 3) Conv branch as implicit-GEMM WMMA -> CP
    conv3x3_gemm_wmma<<<dim3(1, (unsigned)((M + 127) / 128)), 256, 0, stream>>>(
        Y, conv2d_w, conv2d_b, CP, B, Hh, Ww);
    // 4) LN(cn) + GELU -> YHAT columns [192,256)
    ln_kernel<<<(int)((M + 3) / 4), 128, 0, stream>>>(CP, 64, 0, cn_g, cn_b,
                                                      YHAT, 256, 192, 64, (int)M, 1);
    // 5) SR1: dw 8x8/8 on Y[:, 64:] -> X1, LN+GELU in place
    {
        long tot = (long)B * 49 * 192;
        dwconv_kernel<<<(int)((tot + 255) / 256), 256, 0, stream>>>(
            Y, 256, 64, sr1_w, sr1_b, X1, 192, 0,
            B, 192, Hh, Ww, 8, 8, 0, 7, 7, 0, 0);
        ln_kernel<<<(B * 49 + 3) / 4, 128, 0, stream>>>(X1, 192, 0, an1_g, an1_b,
                                                        X1, 192, 0, 192, B * 49, 1);
    }
    // 6) SR2: dw 4x4/4 -> X2, LN+GELU in place
    {
        long tot = (long)B * 196 * 192;
        dwconv_kernel<<<(int)((tot + 255) / 256), 256, 0, stream>>>(
            Y, 256, 64, sr2_w, sr2_b, X2, 192, 0,
            B, 192, Hh, Ww, 4, 4, 0, 14, 14, 0, 0);
        ln_kernel<<<(B * 196 + 3) / 4, 128, 0, stream>>>(X2, 192, 0, an2_g, an2_b,
                                                         X2, 192, 0, 192, B * 196, 1);
    }
    // 7) q = ap @ q_w^T  (WMMA, 128x64 tiles: Nc=192)
    gemm_wmma<2,2,4,2><<<dim3(3, (unsigned)((M + 127) / 128)), 256, 0, stream>>>(
        Y, 256, 64, q_w, nullptr, nullptr, 0, Qb, 192, 0, (int)M, 192, 192);
    // 8) kv GEMMs (WMMA, 128x64)
    gemm_wmma<2,2,4,2><<<dim3(3, (B * 49 + 127) / 128), 256, 0, stream>>>(
        X1, 192, 0, kv1_w, nullptr, nullptr, 0, KV1, 192, 0, B * 49, 192, 192);
    gemm_wmma<2,2,4,2><<<dim3(3, (B * 196 + 127) / 128), 256, 0, stream>>>(
        X2, 192, 0, kv2_w, nullptr, nullptr, 0, KV2, 192, 0, B * 196, 192, 192);
    // 9) Attention -> YHAT[:, :96] and [:, 96:192]
    attn_kernel<<<dim3((unsigned)((M + 255) / 256), 4), 256, 0, stream>>>(
        Qb, KV1, YHAT, B, N, 49, 0, 0);
    attn_kernel<<<dim3((unsigned)((M + 255) / 256), 4), 256, 0, stream>>>(
        Qb, KV2, YHAT, B, N, 196, 96, 96);
    // 10) proj + bias + residual(X) -> XR  (WMMA, 128x128 tiles)
    gemm_wmma<4,2,2,4><<<dim3(2, (unsigned)((M + 127) / 128)), 256, 0, stream>>>(
        YHAT, 256, 0, proj_w, proj_b, X, 256, XR, 256, 0, (int)M, 256, 256);
    // 11) LayerNorm n2 -> Y (reuse)
    ln_kernel<<<(int)((M + 3) / 4), 128, 0, stream>>>(XR, 256, 0, n2_g, n2_b,
                                                      Y, 256, 0, 256, (int)M, 0);
    // 12) fc1 + bias -> Hb  (WMMA, 128x128)
    gemm_wmma<4,2,2,4><<<dim3(8, (unsigned)((M + 127) / 128)), 256, 0, stream>>>(
        Y, 256, 0, fc1_w, fc1_b, nullptr, 0, Hb, 1024, 0, (int)M, 1024, 256);
    // 13) MLP depthwise 3x3 + bias, then GELU -> Gb
    {
        long tot = M * 1024;
        dwconv_kernel<<<(int)((tot + 255) / 256), 256, 0, stream>>>(
            Hb, 1024, 0, dw_w, dw_b, Gb, 1024, 0,
            B, 1024, Hh, Ww, 3, 1, 1, Hh, Ww, 0, 1);
    }
    // 14) fc2 + bias + residual(XR) -> out  (WMMA, 128x128)
    gemm_wmma<4,2,2,4><<<dim3(2, (unsigned)((M + 127) / 128)), 256, 0, stream>>>(
        Gb, 1024, 0, fc2_w, fc2_b, XR, 256, out, 256, 0, (int)M, 256, 1024);
}
